// FullModel_41669772705931
// MI455X (gfx1250) — matrix operand order repaired and checked by reading, not compiled, not run on hardware
//
#include <hip/hip_runtime.h>
#include <hip/hip_bf16.h>

// ---------------------------------------------------------------------------
// MoE CNN (classifier + 4 experts) on gfx1250.
// Convs: implicit-GEMM over V_WMMA_F32_16X16X32_F16 (f32 accumulation).
// Block tile 128x64, 8 waves (4Mx2N), 32x32 per wave (4 accumulators).
// B (weight) tile staged via GLOBAL_LOAD_ASYNC_TO_LDS_B128 (ASYNCcnt).
// ---------------------------------------------------------------------------

typedef __attribute__((ext_vector_type(16))) _Float16 v16h;
typedef __attribute__((ext_vector_type(8)))  float    v8f;

union V16F { v16h v; uint4 q[2]; _Float16 h[16]; };
union V8F  { uint4 q; _Float16 h[8]; };

// -------------------- fp32 -> f16 convert (pack) ---------------------------
__global__ void f32_to_f16_kernel(const float* __restrict__ s,
                                  _Float16* __restrict__ d, int n) {
    int i = blockIdx.x * 256 + threadIdx.x;
    if (i < n) d[i] = (_Float16)s[i];
}

// ---------- weight pack: [K,COUT] fp32 -> transposed [COUT,Kpad] f16 -------
__global__ void pack_wT_kernel(const float* __restrict__ w,
                               _Float16* __restrict__ wT,
                               int K, int Kpad, int COUT) {
    int i = blockIdx.x * 256 + threadIdx.x;
    if (i >= COUT * Kpad) return;
    int n = i / Kpad, k = i % Kpad;
    wT[i] = (k < K) ? (_Float16)w[(size_t)k * COUT + n] : (_Float16)0.0f;
}

// -------------------- implicit-GEMM conv, WMMA f16 -------------------------
// src: [Bn, IH, IW, CIN] f16 (NHWC)   wgt: [COUT, Kpad] f16 (transposed, padded)
// dst: [Bn*OH*OW, COUT] f16 NHWC.     bias fp32, ReLU epilogue.
template<int CIN, int COUT>
__global__ __launch_bounds__(256)
void conv_wmma_kernel(const _Float16* __restrict__ src,
                      const _Float16* __restrict__ wgt,
                      const float*    __restrict__ bias,
                      _Float16*       __restrict__ dst,
                      int IH, int IW, int OH, int OW)
{
    constexpr int K    = 9 * CIN;
    constexpr int Kpad = (K + 31) & ~31;
    constexpr int KT   = 32;        // K per WMMA issue (f16)
    constexpr int MT   = 128;
    constexpr int NT   = 64;
    constexpr int LP   = KT + 8;    // row pad: 80B stride keeps 16B alignment

    __shared__ _Float16 sA[MT][LP];   // [m][k]
    __shared__ _Float16 sB[NT][LP];   // [n][k]  (weights already transposed)

    const int tid  = threadIdx.x;
    const int lane = tid & 31;
    const int wave = tid >> 5;
    const int wm   = wave & 3;        // 0..3 -> 32-row M sub-tile
    const int wn   = wave >> 2;       // 0..1 -> 32-col N sub-tile
    const int m0   = blockIdx.x * MT;
    const int n0   = blockIdx.y * NT;

    // A staging: row = tid>>1 (128 rows), 16 halves at (tid&1)*16
    const int arow = tid >> 1;
    const int akc  = (tid & 1) * 16;
    // B staging: row = tid>>2 (64 rows), 8 halves at (tid&3)*8
    const int brow = tid >> 2;
    const int bkc  = (tid & 3) * 8;

    // decode this thread's output pixel once (constant across K loop)
    const int m  = m0 + arow;
    const int bI = m / (OH * OW);
    const int r2 = m % (OH * OW);
    const int oy = r2 / OW;
    const int ox = r2 % OW;

    // LDS byte offset for this thread's async B-tile destination.
    // Generic pointers to LDS carry the LDS offset in addr[31:0] (ISA 10.2).
    const unsigned sB_lds = (unsigned)(unsigned long long)&sB[brow][bkc];
    const unsigned bRowOff = (unsigned)((size_t)(n0 + brow) * Kpad + bkc) * 2u;

    v8f acc00 = {}, acc01 = {}, acc10 = {}, acc11 = {};

    for (int k0 = 0; k0 < Kpad; k0 += KT) {
        // ---- stage A tile (implicit im2col), two b128 per thread ----
        if constexpr (CIN >= 16) {
            // 16 consecutive k share one (kh,kw); channels contiguous+aligned
            const int kaB = k0 + akc;
            const int ci0 = kaB % CIN;
            const int hw  = kaB / CIN;
            const int kw  = hw % 3;
            const int kh  = hw / 3;
            const int iy  = oy * 2 + kh;    // SAME: pad_begin = 0
            const int ix  = ox * 2 + kw;
            uint4 av0 = make_uint4(0u, 0u, 0u, 0u);
            uint4 av1 = make_uint4(0u, 0u, 0u, 0u);
            if (iy < IH && ix < IW) {
                const _Float16* p =
                    src + (((size_t)bI * IH + iy) * IW + ix) * CIN + ci0;
                av0 = *(const uint4*)p;
                av1 = *(const uint4*)(p + 8);
            }
            *(uint4*)&sA[arow][akc]     = av0;
            *(uint4*)&sA[arow][akc + 8] = av1;
        } else {
            V8F t0, t1;
            #pragma unroll
            for (int j = 0; j < 16; ++j) {
                const int ka = k0 + akc + j;
                _Float16 v = (_Float16)0.0f;
                if (ka < K) {
                    const int ci = ka % CIN;
                    const int hw = ka / CIN;
                    const int kw = hw % 3;
                    const int kh = hw / 3;
                    const int iy = oy * 2 + kh;
                    const int ix = ox * 2 + kw;
                    if (iy < IH && ix < IW)
                        v = src[(((size_t)bI * IH + iy) * IW + ix) * CIN + ci];
                }
                if (j < 8) t0.h[j] = v; else t1.h[j - 8] = v;
            }
            *(uint4*)&sA[arow][akc]     = t0.q;
            *(uint4*)&sA[arow][akc + 8] = t1.q;
        }
        // ---- stage B tile: async global->LDS copy (ASYNCcnt) ----
        {
            const unsigned goff = bRowOff + (unsigned)k0 * 2u;
            asm volatile("global_load_async_to_lds_b128 %0, %1, %2"
                         :
                         : "v"(sB_lds), "v"(goff), "s"(wgt)
                         : "memory");
        }
        asm volatile("s_wait_asynccnt 0x0" ::: "memory");
        __syncthreads();

        // ---- fragment fill, all ds_load_b128 (documented f16 layouts) ----
        // A 16x32: lanes 0-15 row=lane, K 0-7 & 16-23; lanes 16-31 K 8-15 & 24-31
        V16F a0, a1, b0, b1;
        const int ar0 = wm * 32 + (lane & 15);
        const int acb = (lane >> 4) * 8;
        a0.q[0] = *(const uint4*)&sA[ar0][acb];
        a0.q[1] = *(const uint4*)&sA[ar0][acb + 16];
        a1.q[0] = *(const uint4*)&sA[ar0 + 16][acb];
        a1.q[1] = *(const uint4*)&sA[ar0 + 16][acb + 16];
        // B 32x16 (as K x N): lane col, lanes 0-15 K 0-7 & 16-23; lanes 16-31 K 8-15 & 24-31
        const int bc0 = wn * 32 + (lane & 15);
        const int bkb = (lane >> 4) * 8;
        b0.q[0] = *(const uint4*)&sB[bc0][bkb];
        b0.q[1] = *(const uint4*)&sB[bc0][bkb + 16];
        b1.q[0] = *(const uint4*)&sB[bc0 + 16][bkb];
        b1.q[1] = *(const uint4*)&sB[bc0 + 16][bkb + 16];

        acc00 = __builtin_amdgcn_wmma_f32_16x16x32_f16(
                    false, a0.v, false, b0.v, (short)0, acc00, false, false);
        acc01 = __builtin_amdgcn_wmma_f32_16x16x32_f16(
                    false, a0.v, false, b1.v, (short)0, acc01, false, false);
        acc10 = __builtin_amdgcn_wmma_f32_16x16x32_f16(
                    false, a1.v, false, b0.v, (short)0, acc10, false, false);
        acc11 = __builtin_amdgcn_wmma_f32_16x16x32_f16(
                    false, a1.v, false, b1.v, (short)0, acc11, false, false);
        __syncthreads();
    }

    // ---- epilogue: bias + ReLU, store f16 NHWC ----
    // C/D layout: VGPR r: lanes 0-15 -> (M=r, N=lane); lanes 16-31 -> (M=r+8)
    const int rb0  = m0 + wm * 32 + ((lane >> 4) * 8);
    const int col0 = n0 + wn * 32 + (lane & 15);
    const float bv0 = bias[col0];
    const float bv1 = bias[col0 + 16];
    #pragma unroll
    for (int r = 0; r < 8; ++r) {
        float v;
        v = acc00[r] + bv0; v = v > 0.0f ? v : 0.0f;
        dst[(size_t)(rb0 + r) * COUT + col0]           = (_Float16)v;
        v = acc01[r] + bv1; v = v > 0.0f ? v : 0.0f;
        dst[(size_t)(rb0 + r) * COUT + col0 + 16]      = (_Float16)v;
        v = acc10[r] + bv0; v = v > 0.0f ? v : 0.0f;
        dst[(size_t)(rb0 + 16 + r) * COUT + col0]      = (_Float16)v;
        v = acc11[r] + bv1; v = v > 0.0f ? v : 0.0f;
        dst[(size_t)(rb0 + 16 + r) * COUT + col0 + 16] = (_Float16)v;
    }
}

// -------------------- global average pool ----------------------------------
// h: [Bn, P, C] f16 -> pooled: [Bn, C] f32.  blockDim.x == C (256).
__global__ void pool_kernel(const _Float16* __restrict__ h,
                            float* __restrict__ pooled, int P, int C) {
    const int b = blockIdx.x;
    const int c = threadIdx.x;
    const _Float16* p = h + (size_t)b * P * C + c;
    float s = 0.0f;
    for (int i = 0; i < P; ++i) s += (float)p[(size_t)i * C];
    pooled[b * C + c] = s / (float)P;
}

// -------------------- head: dense + softmax + routing ----------------------
// pooled: [5][16][256] f32 (net 0 = classifier, 1..4 = experts)
// out: class_probs [16][4] then regressed [16][2], flat fp32.
__global__ void head_kernel(const float* __restrict__ pooled,
                            const float* __restrict__ cWd,   // [256,4]
                            const float* __restrict__ cbd,   // [4]
                            const float* __restrict__ rWd,   // [4,256,2]
                            const float* __restrict__ rbd,   // [4,2]
                            float* __restrict__ out) {
    const int b = threadIdx.x;
    if (b >= 16) return;
    const float* f = pooled + b * 256;
    float logit[4];
    #pragma unroll
    for (int e = 0; e < 4; ++e) {
        float s = cbd[e];
        for (int c = 0; c < 256; ++c) s += f[c] * cWd[c * 4 + e];
        logit[e] = s;
    }
    float mx = logit[0]; int am = 0;
    #pragma unroll
    for (int e = 1; e < 4; ++e) if (logit[e] > mx) { mx = logit[e]; am = e; }
    float ex[4], den = 0.0f;
    #pragma unroll
    for (int e = 0; e < 4; ++e) { ex[e] = __expf(logit[e] - mx); den += ex[e]; }
    #pragma unroll
    for (int e = 0; e < 4; ++e) out[b * 4 + e] = ex[e] / den;

    const float* fe = pooled + (size_t)(am + 1) * 16 * 256 + b * 256;
    #pragma unroll
    for (int r = 0; r < 2; ++r) {
        float s = rbd[am * 2 + r];
        for (int c = 0; c < 256; ++c) s += fe[c] * rWd[((size_t)am * 256 + c) * 2 + r];
        out[64 + b * 2 + r] = s;
    }
}

// ---------------------------------------------------------------------------
extern "C" void kernel_launch(void* const* d_in, const int* in_sizes, int n_in,
                              void* d_out, int out_size, void* d_ws, size_t ws_size,
                              hipStream_t stream) {
    (void)in_sizes; (void)n_in; (void)out_size; (void)ws_size;

    // dims
    constexpr int Bn = 16;
    constexpr int IH1 = 224, OH1 = 112, C0 = 3,   C1 = 64;
    constexpr int IH2 = 112, OH2 = 56,  C2 = 128;
    constexpr int IH3 = 56,  OH3 = 28,  C3 = 256;
    constexpr int M1 = Bn * OH1 * OH1;         // 200704
    constexpr int M2 = Bn * OH2 * OH2;         // 50176
    constexpr int M3 = Bn * OH3 * OH3;         // 12544
    constexpr int K1 = 9 * C0, K2 = 9 * C1, K3 = 9 * C2;        // 27, 576, 1152
    constexpr int K1p = (K1 + 31) & ~31;                        // 32
    constexpr int K2p = K2, K3p = K3;                           // mult of 32

    // inputs (setup_inputs order)
    const float* x   = (const float*)d_in[0];
    const float* cW1 = (const float*)d_in[1];  const float* cb1 = (const float*)d_in[2];
    const float* cW2 = (const float*)d_in[3];  const float* cb2 = (const float*)d_in[4];
    const float* cW3 = (const float*)d_in[5];  const float* cb3 = (const float*)d_in[6];
    const float* cWd = (const float*)d_in[7];  const float* cbd = (const float*)d_in[8];
    const float* rW1 = (const float*)d_in[9];  const float* rb1 = (const float*)d_in[10];
    const float* rW2 = (const float*)d_in[11]; const float* rb2 = (const float*)d_in[12];
    const float* rW3 = (const float*)d_in[13]; const float* rb3 = (const float*)d_in[14];
    const float* rWd = (const float*)d_in[15]; const float* rbd = (const float*)d_in[16];

    // workspace carve-up (256B aligned)
    char*  ws  = (char*)d_ws;
    size_t off = 0;
    auto carve = [&](size_t bytes) -> char* {
        char* p = ws + off;
        off = (off + bytes + 255) & ~(size_t)255;
        return p;
    };
    _Float16* xh  = (_Float16*)carve((size_t)Bn * IH1 * IH1 * C0 * 2);
    _Float16* w1t = (_Float16*)carve((size_t)5 * C1 * K1p * 2);
    _Float16* w2t = (_Float16*)carve((size_t)5 * C2 * K2p * 2);
    _Float16* w3t = (_Float16*)carve((size_t)5 * C3 * K3p * 2);
    _Float16* h1  = (_Float16*)carve((size_t)M1 * C1 * 2);
    _Float16* h2  = (_Float16*)carve((size_t)M2 * C2 * 2);
    _Float16* h3  = (_Float16*)carve((size_t)M3 * C3 * 2);
    float*    pooled = (float*)carve((size_t)5 * Bn * C3 * 4);

    // ---- convert input to f16 ----
    {
        int n = Bn * IH1 * IH1 * C0;
        f32_to_f16_kernel<<<(n + 255) / 256, 256, 0, stream>>>(x, xh, n);
    }
    // ---- pack+transpose+pad all conv weights: [K,COUT] -> [COUT,Kpad] ------
    auto packT = [&](const float* w, _Float16* wT, int K, int Kp, int CO) {
        int n = CO * Kp;
        pack_wT_kernel<<<(n + 255) / 256, 256, 0, stream>>>(w, wT, K, Kp, CO);
    };
    for (int z = 0; z < 5; ++z) {
        const int e = z - 1;
        packT(z == 0 ? cW1 : rW1 + (size_t)e * K1 * C1, w1t + (size_t)z * C1 * K1p, K1, K1p, C1);
        packT(z == 0 ? cW2 : rW2 + (size_t)e * K2 * C2, w2t + (size_t)z * C2 * K2p, K2, K2p, C2);
        packT(z == 0 ? cW3 : rW3 + (size_t)e * K3 * C3, w3t + (size_t)z * C3 * K3p, K3, K3p, C3);
    }

    // ---- 5 networks sequentially (z=0 classifier, z=1..4 experts) ----------
    for (int z = 0; z < 5; ++z) {
        const int e = z - 1;
        const float* b1 = (z == 0) ? cb1 : rb1 + e * C1;
        const float* b2 = (z == 0) ? cb2 : rb2 + e * C2;
        const float* b3 = (z == 0) ? cb3 : rb3 + e * C3;

        conv_wmma_kernel<C0, C1><<<dim3(M1 / 128, C1 / 64), 256, 0, stream>>>(
            xh, w1t + (size_t)z * C1 * K1p, b1, h1, IH1, IH1, OH1, OH1);
        conv_wmma_kernel<C1, C2><<<dim3(M2 / 128, C2 / 64), 256, 0, stream>>>(
            h1, w2t + (size_t)z * C2 * K2p, b2, h2, IH2, IH2, OH2, OH2);
        conv_wmma_kernel<C2, C3><<<dim3(M3 / 128, C3 / 64), 256, 0, stream>>>(
            h2, w3t + (size_t)z * C3 * K3p, b3, h3, IH3, IH3, OH3, OH3);
        pool_kernel<<<Bn, C3, 0, stream>>>(h3, pooled + (size_t)z * Bn * C3,
                                           OH3 * OH3, C3);
    }

    // ---- head: classifier softmax, argmax routing, expert select -----------
    head_kernel<<<1, 32, 0, stream>>>(pooled, cWd, cbd, rWd, rbd, (float*)d_out);
}
